// MotionVAE_55087250538764
// MI455X (gfx1250) — compile-verified, matrix operand order checked by reading, not run on hardware
//
#include <hip/hip_runtime.h>
#include <hip/hip_bf16.h>

typedef _Float16 half_t;
typedef __attribute__((ext_vector_type(16))) _Float16 v16h;
typedef __attribute__((ext_vector_type(8)))  _Float16 v8h;
typedef __attribute__((ext_vector_type(8)))  float    v8f;

#define BATCH 65536
#define ROWS  64
#define NBLK  (BATCH / ROWS)

// d_out layout (floats): out[B*256], z[B*64], mu[B*64], logvar[B*64]
#define OUT_OFF 0
#define Z_OFF   (BATCH * 256)
#define MU_OFF  (BATCH * 256 + BATCH * 64)
#define LV_OFF  (BATCH * 256 + 2 * BATCH * 64)

// ---- workspace layout (in half_t elements), all K-slice-major: [K/32][N][32]
#define WS_ENC1 0u          // 512x256 -> 131072
#define WS_ENC2 131072u     // 256x256 -> 65536
#define WS_MU   196608u     // 256x64  -> 16384
#define WS_LV   212992u     // 256x64  -> 16384
#define WS_G1   229376u     // 320x64  -> 20480
#define WS_G2   249856u     // 64x64   -> 4096
#define WS_DEC0 253952u     // 1920x256 -> 491520 each
#define WS_DEC1 745472u
#define WS_DEC2 1236992u
#define WS_DEC3 1728512u

// ---- LDS layout (bytes)
#define SM_XCAT   0u        // half [64][520]  (66560 B) ; reused as float out[64][260]
#define SM_X0     66560u    // half [64][328]  (41984 B)
#define SM_XIN    108544u   // half [64][328]  (41984 B)
#define SM_WSLC   150528u   // half [2][256][40] (40960 B)
#define SM_MU     191488u   // half [64][72]   (9216 B)
#define SM_GACT   200704u   // half [2][64][72] (18432 B)
#define SM_COEF   219136u   // float [64][8]   (2048 B)
#define SM_RED    221184u   // float [2][64][4] (2048 B)
#define SMEM_BYTES 223232u

// Scheduling-hint macro (zero-risk: compiled out if builtin missing)
#if defined(__has_builtin)
#if __has_builtin(__builtin_amdgcn_sched_group_barrier)
#define SCHED_GROUP(m, n, id) __builtin_amdgcn_sched_group_barrier((m), (n), (id))
#else
#define SCHED_GROUP(m, n, id)
#endif
#else
#define SCHED_GROUP(m, n, id)
#endif

// fast ELU via hardware v_exp_f32
__device__ __forceinline__ float eluf(float x) {
  return x > 0.0f ? x : (__expf(x) - 1.0f);
}

// Load one wave32 WMMA fragment (A or B operand) given a per-lane row base
// pointer. Per ISA 16-bit layout: lane<16 holds K octets {0..7,16..23},
// lane>=16 holds {8..15,24..31}; both are two contiguous 16B chunks.
__device__ __forceinline__ v16h load_frag16(const half_t* base, int g) {
  v8h lo = *(const v8h*)(base + g * 8);
  v8h hi = *(const v8h*)(base + 16 + g * 8);
  v16h r;
#pragma unroll
  for (int i = 0; i < 8; ++i) { r[i] = lo[i]; r[i + 8] = hi[i]; }
  return r;
}

__device__ __forceinline__ v8f wmma_f16(v16h a, v16h b, v8f c) {
  return __builtin_amdgcn_wmma_f32_16x16x32_f16(false, a, false, b, (short)0, c,
                                                false, false);
}

// LDS-staged GEMM: C[64x256] += A[64 x 32*OUTER*WRAP] * W.
// Weights wg are K-slice-major [KS][256][32] halves in global memory (L2-hot).
// K-slices stream into a double-buffered LDS tile with the CDNA5 async
// global->LDS DMA path (ASYNCcnt); one barrier per K-iteration suffices with
// double buffering when the barrier sits at the loop top. Each wave owns
// m-tile (wave&3) and 8 n-tiles. Scheduling groups force all 18 DS reads
// (A + 8xB fragments) ahead of the 8 WMMAs so dscnt waits are partial and
// the WMMAs issue back-to-back. SCALED folds the MoE softmax coefficient
// into A per expert block (OUTER = E).
template <int OUTER, int WRAP, bool SCALED>
__device__ void gemm_staged(const half_t* __restrict__ wg, const half_t* sA,
                            int strideA, const float* sCoef, half_t* swslc,
                            v8f (&acc)[8]) {
  constexpr int KS = OUTER * WRAP;
  const int tid = threadIdx.x;
  const int lane = tid & 31;
  const int wave = tid >> 5;
  const int g = lane >> 4, r = lane & 15;
  const int m = wave & 3, nh = wave >> 2;
  const half_t* arow = sA + (m * 16 + r) * strideA;
  // wave-relative LDS byte address of this thread's staging row
  const unsigned lds_row = (unsigned)(size_t)(const void*)swslc + (unsigned)tid * 80u;
  const int grow = tid * 64;  // byte offset of this thread's row in a slice

  // preload K-slice 0 into buffer 0 via async DMA
#pragma unroll
  for (int c = 0; c < 4; ++c) {
    unsigned la = lds_row + c * 16u;
    int ga = grow + c * 16;
    asm volatile("global_load_async_to_lds_b128 %0, %1, %2"
                 :: "v"(la), "v"(ga), "s"((const void*)wg) : "memory");
  }
  asm volatile("s_wait_asynccnt 0x0" ::: "memory");

  int buf = 0;
#pragma unroll 1
  for (int e = 0; e < OUTER; ++e) {
    half_t csc = (half_t)1.0f;
    if (SCALED) csc = (half_t)sCoef[(m * 16 + r) * 8 + e];
#pragma unroll 1
    for (int k2 = 0; k2 < WRAP; ++k2) {
      const int kk = e * WRAP + k2;
      __syncthreads();  // publish buffer `buf`; WAR-protect buffer buf^1
      if (kk + 1 < KS) {  // async-stream next slice, overlapped with compute
#pragma unroll
        for (int c = 0; c < 4; ++c) {
          unsigned la = lds_row + (unsigned)(buf ^ 1) * 20480u + c * 16u;
          int ga = (kk + 1) * 16384 + grow + c * 16;
          asm volatile("global_load_async_to_lds_b128 %0, %1, %2"
                       :: "v"(la), "v"(ga), "s"((const void*)wg) : "memory");
        }
      }
      v16h a = load_frag16(arow + k2 * 32, g);
      if (SCALED) {
#pragma unroll
        for (int i = 0; i < 16; ++i) a[i] *= csc;
      }
      const half_t* wb = swslc + buf * (256 * 40) + (nh * 128 + r) * 40;
      v16h bf[4];
#pragma unroll
      for (int j = 0; j < 4; ++j) bf[j] = load_frag16(wb + j * 640, g);
#pragma unroll
      for (int j = 0; j < 4; ++j) acc[j] = wmma_f16(a, bf[j], acc[j]);
#pragma unroll
      for (int j = 0; j < 4; ++j) bf[j] = load_frag16(wb + (j + 4) * 640, g);
#pragma unroll
      for (int j = 0; j < 4; ++j) acc[j + 4] = wmma_f16(a, bf[j], acc[j + 4]);
      // scheduling groups: all DS reads first, then the 8 WMMAs
      SCHED_GROUP(0x100, 18, 0);  // DS read x18 (A frag + 8 B frags)
      SCHED_GROUP(0x008, 8, 0);   // WMMA x8
      if (kk + 1 < KS) {
        asm volatile("s_wait_asynccnt 0x0" ::: "memory");  // our DMA done
        buf ^= 1;
      }
    }
  }
}

// Small GEMM tile (NOUT=64), B fragments loaded directly from global (L2-hot).
template <int KS>
__device__ v8f gemm_direct_tile(const half_t* __restrict__ wg, const half_t* sA,
                                int strideA, int mt, int nt, int g, int r) {
  v8f acc = {};
  const half_t* arow = sA + (mt * 16 + r) * strideA;
#pragma unroll
  for (int kk = 0; kk < KS; ++kk) {
    v16h a = load_frag16(arow + kk * 32, g);
    v16h b = load_frag16(wg + ((size_t)kk * 64 + nt * 16 + r) * 32, g);
    acc = wmma_f16(a, b, acc);
  }
  return acc;
}

// Convert f32 [din,dout] weights to f16 in K-slice-major layout [din/32][dout][32].
__global__ void prep_w(const float* __restrict__ src, half_t* __restrict__ dst,
                       int din, int dout) {
  int idx = blockIdx.x * 256 + threadIdx.x;
  if (idx >= din * dout) return;
  int k = idx / dout, n = idx % dout;
  int kk = k >> 5, kr = k & 31;
  dst[((size_t)(kk * dout + n)) * 32 + kr] = (half_t)src[idx];
}

__global__ void __launch_bounds__(256) motion_vae_kernel(
    const float* __restrict__ curr, const float* __restrict__ next,
    const float* __restrict__ eps, const float* __restrict__ enc_b1,
    const float* __restrict__ enc_b2, const float* __restrict__ mu_b,
    const float* __restrict__ lv_b, const float* __restrict__ g_b1v,
    const float* __restrict__ g_b2v, const float* __restrict__ g_w3,
    const float* __restrict__ g_b3, const float* __restrict__ bd0,
    const float* __restrict__ bd1, const float* __restrict__ bd2,
    const float* __restrict__ bd3, const float* __restrict__ ln0s,
    const float* __restrict__ ln0b, const float* __restrict__ ln1s,
    const float* __restrict__ ln1b, const float* __restrict__ ln2s,
    const float* __restrict__ ln2b, const float* __restrict__ ln3s,
    const float* __restrict__ ln3b, const half_t* __restrict__ ws,
    float* __restrict__ out) {
  extern __shared__ char smem[];
  half_t* s_xcat = (half_t*)(smem + SM_XCAT);  // [64][520]
  float*  s_out  = (float*)(smem + SM_XCAT);   // [64][260] (reuses xcat space)
  half_t* s_x0   = (half_t*)(smem + SM_X0);    // [64][328]
  half_t* s_xin  = (half_t*)(smem + SM_XIN);   // [64][328]
  half_t* s_wslc = (half_t*)(smem + SM_WSLC);  // [2][256][40]
  half_t* s_mu   = (half_t*)(smem + SM_MU);    // [64][72]
  half_t* s_gact = (half_t*)(smem + SM_GACT);  // [2][64][72]
  float*  s_coef = (float*)(smem + SM_COEF);   // [64][8]
  float*  s_red  = (float*)(smem + SM_RED);    // [2][64][4]

  const int tid = threadIdx.x;
  const int lane = tid & 31;
  const int wave = tid >> 5;
  const int g = lane >> 4, r = lane & 15;
  const int m4 = wave & 3, nh = wave >> 2;
  const int r0 = blockIdx.x * ROWS;

  // ---- stage 0: xcat = concat(curr, next) -> f16 LDS
  for (int i = tid; i < 64 * 512; i += 256) {
    int rr = i >> 9, c = i & 511;
    float v = (c < 256) ? curr[(size_t)(r0 + rr) * 256 + c]
                        : next[(size_t)(r0 + rr) * 256 + (c - 256)];
    s_xcat[rr * 520 + c] = (half_t)v;
  }
  __syncthreads();

  // ---- encoder layer 1: h = elu(xcat @ enc_w1 + b1) -> s_x0
  {
    v8f acc[8] = {};
    gemm_staged<1, 16, false>(ws + WS_ENC1, s_xcat, 520, nullptr, s_wslc, acc);
#pragma unroll
    for (int j = 0; j < 8; ++j) {
      int col = (nh * 8 + j) * 16 + r;
      float bb = enc_b1[col];
#pragma unroll
      for (int el = 0; el < 8; ++el) {
        int rr = m4 * 16 + g * 8 + el;
        s_x0[rr * 328 + col] = (half_t)eluf(acc[j][el] + bb);
      }
    }
  }
  __syncthreads();

  // ---- encoder layer 2: h2 = elu(h @ enc_w2 + b2) -> s_xin
  {
    v8f acc[8] = {};
    gemm_staged<1, 8, false>(ws + WS_ENC2, s_x0, 328, nullptr, s_wslc, acc);
#pragma unroll
    for (int j = 0; j < 8; ++j) {
      int col = (nh * 8 + j) * 16 + r;
      float bb = enc_b2[col];
#pragma unroll
      for (int el = 0; el < 8; ++el) {
        int rr = m4 * 16 + g * 8 + el;
        s_xin[rr * 328 + col] = (half_t)eluf(acc[j][el] + bb);
      }
    }
  }
  __syncthreads();

  // ---- mu / logvar / z (N=64; 16 tiles, 2 per wave; dual accumulators)
#pragma unroll
  for (int ti = 0; ti < 2; ++ti) {
    int t = wave * 2 + ti;
    int mt = t & 3, nt = t >> 2;
    v8f am = gemm_direct_tile<8>(ws + WS_MU, s_xin, 328, mt, nt, g, r);
    v8f al = gemm_direct_tile<8>(ws + WS_LV, s_xin, 328, mt, nt, g, r);
    int col = nt * 16 + r;
#pragma unroll
    for (int el = 0; el < 8; ++el) {
      int rr = mt * 16 + g * 8 + el;
      int gi = (r0 + rr) * 64 + col;
      float mv = am[el] + mu_b[col];
      float lv = al[el] + lv_b[col];
      out[MU_OFF + gi] = mv;
      out[LV_OFF + gi] = lv;
      out[Z_OFF + gi] = mv + eps[gi] * __expf(0.5f * lv);
      s_mu[rr * 72 + col] = (half_t)mv;
    }
  }
  __syncthreads();

  // ---- x0 = concat(mu, curr) (f16) -> s_x0 (h is dead)
  for (int i = tid; i < 64 * 320; i += 256) {
    int rr = i / 320, c = i % 320;
    s_x0[rr * 328 + c] = (c < 64) ? s_mu[rr * 72 + c] : s_xcat[rr * 520 + (c - 64)];
  }
  __syncthreads();

  // ---- gating layer 1: elu(x0 @ g_w1 + b) -> s_gact[0]
#pragma unroll
  for (int ti = 0; ti < 2; ++ti) {
    int t = wave * 2 + ti;
    int mt = t & 3, nt = t >> 2;
    v8f a = gemm_direct_tile<10>(ws + WS_G1, s_x0, 328, mt, nt, g, r);
    int col = nt * 16 + r;
#pragma unroll
    for (int el = 0; el < 8; ++el) {
      int rr = mt * 16 + g * 8 + el;
      s_gact[rr * 72 + col] = (half_t)eluf(a[el] + g_b1v[col]);
    }
  }
  __syncthreads();

  // ---- gating layer 2 -> s_gact[1]
#pragma unroll
  for (int ti = 0; ti < 2; ++ti) {
    int t = wave * 2 + ti;
    int mt = t & 3, nt = t >> 2;
    v8f a = gemm_direct_tile<2>(ws + WS_G2, s_gact, 72, mt, nt, g, r);
    int col = nt * 16 + r;
#pragma unroll
    for (int el = 0; el < 8; ++el) {
      int rr = mt * 16 + g * 8 + el;
      s_gact[64 * 72 + rr * 72 + col] = (half_t)eluf(a[el] + g_b2v[col]);
    }
  }
  __syncthreads();

  // ---- gating layer 3 + softmax (E=6) -> s_coef
  if (tid < 64) {
    const half_t* grow = s_gact + 64 * 72 + tid * 72;
    float o[6];
#pragma unroll
    for (int e = 0; e < 6; ++e) o[e] = g_b3[e];
    for (int k = 0; k < 64; ++k) {
      float v = (float)grow[k];
#pragma unroll
      for (int e = 0; e < 6; ++e) o[e] += v * g_w3[k * 6 + e];
    }
    float mx = o[0];
#pragma unroll
    for (int e = 1; e < 6; ++e) mx = fmaxf(mx, o[e]);
    float sum = 0.0f;
#pragma unroll
    for (int e = 0; e < 6; ++e) { o[e] = __expf(o[e] - mx); sum += o[e]; }
    float inv = 1.0f / sum;
#pragma unroll
    for (int e = 0; e < 6; ++e) s_coef[tid * 8 + e] = o[e] * inv;
  }
  __syncthreads();

  // ---- decoder: 4 layers of LN + (MoE as one K=1920 GEMM with row scaling)
  const float* lnS[4] = {ln0s, ln1s, ln2s, ln3s};
  const float* lnB[4] = {ln0b, ln1b, ln2b, ln3b};
  const float* bsL[4] = {bd0, bd1, bd2, bd3};
  const unsigned wofs[4] = {WS_DEC0, WS_DEC1, WS_DEC2, WS_DEC3};

#pragma unroll 1
  for (int L = 0; L < 4; ++L) {
    // LayerNorm over 320 cols -> s_xin (f16). 4 threads per row.
    {
      int tq = tid & 3, rr = tid >> 2;
      float su = 0.0f, sq = 0.0f;
      for (int c = tq * 80; c < tq * 80 + 80; ++c) {
        float v = (L == 0) ? (float)s_x0[rr * 328 + c]
                           : (c < 64 ? (float)s_mu[rr * 72 + c]
                                     : s_out[rr * 260 + (c - 64)]);
        su += v;
        sq += v * v;
      }
      s_red[rr * 4 + tq] = su;
      s_red[256 + rr * 4 + tq] = sq;
      __syncthreads();
      float mean = (s_red[rr * 4] + s_red[rr * 4 + 1] + s_red[rr * 4 + 2] +
                    s_red[rr * 4 + 3]) * (1.0f / 320.0f);
      float msq = (s_red[256 + rr * 4] + s_red[256 + rr * 4 + 1] +
                   s_red[256 + rr * 4 + 2] + s_red[256 + rr * 4 + 3]) *
                  (1.0f / 320.0f);
      float rstd = rsqrtf(msq - mean * mean + 1e-5f);
      const float* Lsc = lnS[L];
      const float* Lbi = lnB[L];
      for (int c = tq * 80; c < tq * 80 + 80; ++c) {
        float v = (L == 0) ? (float)s_x0[rr * 328 + c]
                           : (c < 64 ? (float)s_mu[rr * 72 + c]
                                     : s_out[rr * 260 + (c - 64)]);
        s_xin[rr * 328 + c] = (half_t)((v - mean) * rstd * Lsc[c] + Lbi[c]);
      }
    }
    __syncthreads();

    v8f acc[8] = {};
    gemm_staged<6, 10, true>(ws + wofs[L], s_xin, 328, s_coef, s_wslc, acc);

    // epilogue: rows outer so the 6 coefs are loaded once per row and reused
    // across all 8 n-tiles.
    const float* bl = bsL[L];
#pragma unroll
    for (int el = 0; el < 8; ++el) {
      int rr = m4 * 16 + g * 8 + el;
      float c0 = s_coef[rr * 8 + 0], c1 = s_coef[rr * 8 + 1];
      float c2 = s_coef[rr * 8 + 2], c3 = s_coef[rr * 8 + 3];
      float c4 = s_coef[rr * 8 + 4], c5 = s_coef[rr * 8 + 5];
#pragma unroll
      for (int j = 0; j < 8; ++j) {
        int col = (nh * 8 + j) * 16 + r;
        float bb = c0 * bl[col] + c1 * bl[256 + col] + c2 * bl[512 + col] +
                   c3 * bl[768 + col] + c4 * bl[1024 + col] + c5 * bl[1280 + col];
        float mix = acc[j][el] + bb;
        if (L == 0)
          s_out[rr * 260 + col] = eluf(mix);
        else if (L < 3)
          s_out[rr * 260 + col] = eluf(s_out[rr * 260 + col] + mix);
        else
          out[(size_t)(r0 + rr) * 256 + col] = mix;
      }
    }
    __syncthreads();
  }
}

extern "C" void kernel_launch(void* const* d_in, const int* in_sizes, int n_in,
                              void* d_out, int out_size, void* d_ws,
                              size_t ws_size, hipStream_t stream) {
  half_t* ws = (half_t*)d_ws;
  (void)in_sizes; (void)n_in; (void)out_size; (void)ws_size;

  // allow >64KB dynamic LDS (CDNA5 WGP has 320KB)
  hipFuncSetAttribute((const void*)motion_vae_kernel,
                      hipFuncAttributeMaxDynamicSharedMemorySize,
                      (int)SMEM_BYTES);

  auto prep = [&](int idx, unsigned off, int din, int dout) {
    int n = din * dout;
    prep_w<<<(n + 255) / 256, 256, 0, stream>>>((const float*)d_in[idx],
                                                ws + off, din, dout);
  };
  prep(3, WS_ENC1, 512, 256);
  prep(5, WS_ENC2, 256, 256);
  prep(7, WS_MU, 256, 64);
  prep(9, WS_LV, 256, 64);
  prep(11, WS_G1, 320, 64);
  prep(13, WS_G2, 64, 64);
  prep(17, WS_DEC0, 1920, 256);
  prep(21, WS_DEC1, 1920, 256);
  prep(25, WS_DEC2, 1920, 256);
  prep(29, WS_DEC3, 1920, 256);

  motion_vae_kernel<<<NBLK, 256, SMEM_BYTES, stream>>>(
      (const float*)d_in[0], (const float*)d_in[1], (const float*)d_in[2],
      (const float*)d_in[4], (const float*)d_in[6], (const float*)d_in[8],
      (const float*)d_in[10], (const float*)d_in[12], (const float*)d_in[14],
      (const float*)d_in[15], (const float*)d_in[16], (const float*)d_in[18],
      (const float*)d_in[22], (const float*)d_in[26], (const float*)d_in[30],
      (const float*)d_in[19], (const float*)d_in[20], (const float*)d_in[23],
      (const float*)d_in[24], (const float*)d_in[27], (const float*)d_in[28],
      (const float*)d_in[31], (const float*)d_in[32], (const half_t*)d_ws,
      (float*)d_out);
}
